// GAT_51539607552881
// MI455X (gfx1250) — matrix operand order repaired and checked by reading, not compiled, not run on hardware
//
#include <hip/hip_runtime.h>
#include <hip/hip_bf16.h>
#include <stdint.h>

typedef __attribute__((ext_vector_type(16))) _Float16 v16h;
typedef __attribute__((ext_vector_type(8)))  float    v8f;

#define NEG_SLOPE 0.2f

// ---------- monotone float <-> uint encoding for atomicMax on floats ----------
__device__ __forceinline__ unsigned enc_ord(float f) {
  unsigned u = __float_as_uint(f);
  return (u & 0x80000000u) ? ~u : (u | 0x80000000u);
}
__device__ __forceinline__ float dec_ord(unsigned u) {
  u = (u & 0x80000000u) ? (u & 0x7fffffffu) : ~u;
  return __uint_as_float(u);
}

// ---------------------------- utility: zero fill ----------------------------
__global__ void k_zero_u32(unsigned* __restrict__ p, size_t n) {
  size_t i = (size_t)blockIdx.x * blockDim.x + threadIdx.x;
  if (i < n) p[i] = 0u;
}

// ---------------------------------------------------------------------------
// Dense projection: out[n x CA] = A[n x K] * W[K x CA] (+bias) (opt relu)
// f16 WMMA with f32 accumulate. K % 32 == 0, CT % 16 == 0, CA <= CT.
// 256 threads = 8 waves; each wave owns a 16-row tile -> 128 rows per block.
// W is staged in LDS as f16, pre-transposed (Wt[c][k]) so each lane's
// B-fragment is one contiguous 32B LDS read (ds_load_b128 x2).
// ---------------------------------------------------------------------------
template<int K, int CT, int CA>
__global__ __launch_bounds__(256)
void k_gemm_wmma(const float* __restrict__ A, const float* __restrict__ W,
                 const float* __restrict__ bias, float* __restrict__ out,
                 int n, int do_relu)
{
  __shared__ __align__(32) _Float16 Wt[CT * K];
  const int tid = threadIdx.x;
  for (int idx = tid; idx < CT * K; idx += 256) {
    int c = idx / K, k = idx - c * K;
    Wt[idx] = (c < CA) ? (_Float16)W[(size_t)k * CA + c] : (_Float16)0.0f;
  }
  __syncthreads();

  const int wave = tid >> 5;
  const int lane = tid & 31;
  const int nl   = lane & 15;   // N (B/C/D) or M (A) within tile
  const int hh   = lane >> 4;   // lane half selects K-range (A/B) / row half (D)
  const int rowsBase = (blockIdx.x * 8 + wave) * 16;

  int arow = rowsBase + nl;               // A fragment row for this lane
  if (arow > n - 1) arow = n - 1;         // clamp tail (EXEC must stay all-1s)
  const float* Abase = A + (size_t)arow * K;

  v8f acc[CT / 16];
  #pragma unroll
  for (int t = 0; t < CT / 16; t++) acc[t] = (v8f){0, 0, 0, 0, 0, 0, 0, 0};

  #pragma unroll
  for (int kb = 0; kb < K / 32; kb++) {
    // A fragment (16x32 f16): lane holds K = 16h + 0..15 (contiguous)
    v16h a;
    const float* ap = Abase + kb * 32 + hh * 16;
    #pragma unroll
    for (int i = 0; i < 8; i++) a[i] = (_Float16)ap[i];
    #pragma unroll
    for (int i = 0; i < 8; i++) a[8 + i] = (_Float16)ap[8 + i];

    #pragma unroll
    for (int t = 0; t < CT / 16; t++) {
      // B fragment (32x16 f16): lane holds col nl, K = 16h + 0..15 (contiguous)
      const v16h* bp = (const v16h*)&Wt[(size_t)(t * 16 + nl) * K + kb * 32 + hh * 16];
      v16h b = *bp;
      acc[t] = __builtin_amdgcn_wmma_f32_16x16x32_f16(
          false, a, false, b, (short)0, acc[t], false, false);
    }
  }

  // D layout: VGPR r -> row (8*hh + r), col nl within the 16x16 tile
  #pragma unroll
  for (int t = 0; t < CT / 16; t++) {
    int col = t * 16 + nl;
    if (col < CA) {
      float badd = bias ? bias[col] : 0.0f;
      #pragma unroll
      for (int r = 0; r < 8; r++) {
        int row = rowsBase + hh * 8 + r;
        if (row < n) {
          float v = acc[t][r] + badd;
          if (do_relu) v = fmaxf(v, 0.0f);
          out[(size_t)row * CA + col] = v;
        }
      }
    }
  }
}

// ----------------- per-node attention coefficients (wave/node) -----------------
template<int C>
__global__ __launch_bounds__(256)
void k_alpha(const float* __restrict__ h, const float* __restrict__ a_src,
             const float* __restrict__ a_dst, float* __restrict__ As,
             float* __restrict__ Ad, int n)
{
  int node = blockIdx.x * 8 + (threadIdx.x >> 5);
  int lane = threadIdx.x & 31;
  if (node >= n) return;
  float ss = 0.f, dd = 0.f;
  for (int c = lane; c < C; c += 32) {
    float v = h[(size_t)node * C + c];
    ss += v * a_src[c];
    dd += v * a_dst[c];
  }
  #pragma unroll
  for (int m = 16; m > 0; m >>= 1) {
    ss += __shfl_xor(ss, m, 32);
    dd += __shfl_xor(dd, m, 32);
  }
  if (lane == 0) { As[node] = ss; Ad[node] = dd; }
}

// ------------------ edge pass 1: leaky-relu logit + segment max ------------------
__global__ void k_logit_max(const int* __restrict__ src, const int* __restrict__ dst,
                            const float* __restrict__ As, const float* __restrict__ Ad,
                            float* __restrict__ ew, unsigned* __restrict__ menc,
                            int E, int NE)
{
  int e = blockIdx.x * blockDim.x + threadIdx.x;
  if (e >= NE) return;
  int s = (e < E) ? src[e] : (e - E);   // edges then self-loops
  int d = (e < E) ? dst[e] : (e - E);
  float z = As[s] + Ad[d];
  z = (z > 0.f) ? z : z * NEG_SLOPE;
  ew[e] = z;
  atomicMax(&menc[d], enc_ord(z));
}

// ------- edge pass 2 (fused): w = exp(logit-max); denom += w; agg += w*h[src] -------
// wave-per-edge: lanes stride the feature dim -> coalesced gather + coalesced
// atomic adds. Normalization by denom is deferred to the k_finish epilogue.
template<int C>
__global__ __launch_bounds__(256)
void k_exp_aggregate(const int* __restrict__ src, const int* __restrict__ dst,
                     const float* __restrict__ ew, const unsigned* __restrict__ menc,
                     float* __restrict__ denom, const float* __restrict__ h,
                     float* __restrict__ agg, int E, int NE)
{
  int e = blockIdx.x * 8 + (threadIdx.x >> 5);
  int lane = threadIdx.x & 31;
  if (e >= NE) return;
  int s = (e < E) ? src[e] : (e - E);
  int d = (e < E) ? dst[e] : (e - E);
  float w = __expf(ew[e] - dec_ord(menc[d]));
  if (lane == 0) atomicAdd(&denom[d], w);
  for (int c = lane; c < C; c += 32)
    atomicAdd(&agg[(size_t)d * C + c], h[(size_t)s * C + c] * w);
}

// --------------- epilogue: out = relu(agg/denom + bias) per element ---------------
template<int C>
__global__ void k_finish(const float* __restrict__ agg, const float* __restrict__ denom,
                         const float* __restrict__ b, float* __restrict__ out, int n)
{
  size_t i = (size_t)blockIdx.x * blockDim.x + threadIdx.x;
  if (i >= (size_t)n * C) return;
  int node = (int)(i / C);
  int c    = (int)(i % C);
  out[i] = fmaxf(agg[i] / denom[node] + b[c], 0.f);
}

// ---------------------------------------------------------------------------
extern "C" void kernel_launch(void* const* d_in, const int* in_sizes, int n_in,
                              void* d_out, int out_size, void* d_ws, size_t ws_size,
                              hipStream_t stream)
{
  const float* x    = (const float*)d_in[0];
  const int*   ei   = (const int*)  d_in[1];
  const float* W1   = (const float*)d_in[2];
  const float* as1  = (const float*)d_in[3];
  const float* ad1  = (const float*)d_in[4];
  const float* b1   = (const float*)d_in[5];
  const float* W2   = (const float*)d_in[6];
  const float* as2  = (const float*)d_in[7];
  const float* ad2  = (const float*)d_in[8];
  const float* b2   = (const float*)d_in[9];
  const float* fcW  = (const float*)d_in[10];
  const float* fcb  = (const float*)d_in[11];

  const int N  = in_sizes[0] / 256;   // IN_DIM = 256
  const int E  = in_sizes[1] / 2;     // edge_index is [2, E]
  const int NE = E + N;               // + self loops
  const int* srcI = ei;
  const int* dstI = ei + E;

  float* emb    = (float*)d_out;                   // [N, 32]
  float* logits = (float*)d_out + (size_t)N * 32;  // [N, 40]

  // -------- workspace carve --------
  char* w = (char*)d_ws;
  auto carve = [&](size_t bytes) -> void* {
    void* p = (void*)w;
    w += (bytes + 255) & ~(size_t)255;
    return p;
  };
  float*    h1    = (float*)   carve(sizeof(float) * (size_t)N * 64);
  float*    agg1  = (float*)   carve(sizeof(float) * (size_t)N * 64);
  float*    h2    = (float*)   carve(sizeof(float) * (size_t)N * 32);
  float*    agg2  = (float*)   carve(sizeof(float) * (size_t)N * 32);
  float*    As    = (float*)   carve(sizeof(float) * (size_t)N);
  float*    Ad    = (float*)   carve(sizeof(float) * (size_t)N);
  unsigned* menc  = (unsigned*)carve(sizeof(unsigned) * (size_t)N);
  float*    denom = (float*)   carve(sizeof(float) * (size_t)N);
  float*    ew    = (float*)   carve(sizeof(float) * (size_t)NE);

  const int T = 256;
  dim3 blk(T);
  dim3 g_gemm((N + 127) / 128);
  dim3 g_node8((N + 7) / 8);
  dim3 g_edge((NE + T - 1) / T);
  dim3 g_edge8((NE + 7) / 8);

  // ===================== layer 1 =====================
  k_gemm_wmma<256, 64, 64><<<g_gemm, blk, 0, stream>>>(x, W1, nullptr, h1, N, 0);
  k_alpha<64><<<g_node8, blk, 0, stream>>>(h1, as1, ad1, As, Ad, N);

  k_zero_u32<<<dim3((N + T - 1) / T), blk, 0, stream>>>(menc, (size_t)N);
  k_zero_u32<<<dim3((N + T - 1) / T), blk, 0, stream>>>((unsigned*)denom, (size_t)N);
  k_zero_u32<<<dim3(((size_t)N * 64 + T - 1) / T), blk, 0, stream>>>((unsigned*)agg1, (size_t)N * 64);

  k_logit_max<<<g_edge, blk, 0, stream>>>(srcI, dstI, As, Ad, ew, menc, E, NE);
  k_exp_aggregate<64><<<g_edge8, blk, 0, stream>>>(srcI, dstI, ew, menc, denom, h1, agg1, E, NE);
  k_finish<64><<<dim3(((size_t)N * 64 + T - 1) / T), blk, 0, stream>>>(agg1, denom, b1, agg1, N);

  // ===================== layer 2 =====================
  k_gemm_wmma<64, 32, 32><<<g_gemm, blk, 0, stream>>>(agg1, W2, nullptr, h2, N, 0);
  k_alpha<32><<<g_node8, blk, 0, stream>>>(h2, as2, ad2, As, Ad, N);

  k_zero_u32<<<dim3((N + T - 1) / T), blk, 0, stream>>>(menc, (size_t)N);
  k_zero_u32<<<dim3((N + T - 1) / T), blk, 0, stream>>>((unsigned*)denom, (size_t)N);
  k_zero_u32<<<dim3(((size_t)N * 32 + T - 1) / T), blk, 0, stream>>>((unsigned*)agg2, (size_t)N * 32);

  k_logit_max<<<g_edge, blk, 0, stream>>>(srcI, dstI, As, Ad, ew, menc, E, NE);
  k_exp_aggregate<32><<<g_edge8, blk, 0, stream>>>(srcI, dstI, ew, menc, denom, h2, agg2, E, NE);
  k_finish<32><<<dim3(((size_t)N * 32 + T - 1) / T), blk, 0, stream>>>(agg2, denom, b2, emb, N);

  // ===================== FC head (40 cols padded to 48) =====================
  k_gemm_wmma<32, 48, 40><<<g_gemm, blk, 0, stream>>>(emb, fcW, fcb, logits, N, 0);
}
